// KAN_NODE_49563922596369
// MI455X (gfx1250) — matrix operand order repaired and checked by reading, not compiled, not run on hardware
//
#include <hip/hip_runtime.h>

typedef __attribute__((ext_vector_type(16))) __bf16    bf16x16;
typedef __attribute__((ext_vector_type(2)))  __bf16    bf16x2;
typedef __attribute__((ext_vector_type(8)))  float     v8f;
typedef __attribute__((ext_vector_type(8)))  unsigned int u32x8;
typedef __attribute__((ext_vector_type(4)))  unsigned int u32x4;

#define BTOT 2048

#if __has_builtin(__builtin_amdgcn_cvt_pk_bf16_f32)
__device__ __forceinline__ unsigned int pk2(float lo, float hi) {
  bf16x2 p = __builtin_amdgcn_cvt_pk_bf16_f32(lo, hi);
  return __builtin_bit_cast(unsigned int, p);
}
#else
// fptrunc f32->bf16: backend picks the native convert when available.
__device__ __forceinline__ unsigned int pk2(float lo, float hi) {
  bf16x2 p;
  p[0] = (__bf16)lo;
  p[1] = (__bf16)hi;
  return __builtin_bit_cast(unsigned int, p);
}
#endif

__device__ __forceinline__ v8f wmma_bf16(u32x8 a, u32x8 b, v8f c) {
  return __builtin_amdgcn_wmma_f32_16x16x32_bf16(
      false, __builtin_bit_cast(bf16x16, a),
      false, __builtin_bit_cast(bf16x16, b),
      (short)0, c, false, false);
}

// 32B-aligned LDS B-tile fetch as 2 x b128.
__device__ __forceinline__ u32x8 load_btile(const unsigned int* p) {
  const u32x4* q = (const u32x4*)p;
  u32x4 lo = q[0], hi = q[1];
  u32x8 r;
  #pragma unroll
  for (int j = 0; j < 4; ++j) { r[j] = lo[j]; r[4 + j] = hi[j]; }
  return r;
}

__device__ __forceinline__ void wave_lds_sync() {
  asm volatile("s_wait_dscnt 0x0" ::: "memory");
}

// silu via v_rcp_f32 instead of IEEE division.
__device__ __forceinline__ float silu(float x) {
  return x * __builtin_amdgcn_rcpf(1.f + __expf(-x));
}

// ---------------------------------------------------------------------------
// Kernel 0: fold xproj into the ODE input path.
//   M1 = W1x @ xproj_w (128x32), c1 = W1x @ xproj_b + b1
//   M0 = z0_w @ xproj_w (128x32), c0 = z0_w @ xproj_b + z0_b
// ---------------------------------------------------------------------------
__global__ void prep_kernel(const float* __restrict__ w1, const float* __restrict__ b1,
                            const float* __restrict__ xw, const float* __restrict__ xb,
                            const float* __restrict__ z0w, const float* __restrict__ z0b,
                            float* __restrict__ M1, float* __restrict__ c1,
                            float* __restrict__ M0, float* __restrict__ c0) {
  int n = threadIdx.x;  // 128 threads
  for (int d = 0; d < 32; ++d) {
    float s1 = 0.f, s0 = 0.f;
    for (int j = 0; j < 128; ++j) {
      float xv = xw[j * 32 + d];
      s1 += w1[n * 256 + 128 + j] * xv;
      s0 += z0w[n * 128 + j] * xv;
    }
    M1[n * 32 + d] = s1;
    M0[n * 32 + d] = s0;
  }
  float t1 = b1[n], t0 = z0b[n];
  for (int j = 0; j < 128; ++j) {
    t1 += w1[n * 256 + 128 + j] * xb[j];
    t0 += z0w[n * 128 + j] * xb[j];
  }
  c1[n] = t1;
  c0[n] = t0;
}

// ---------------------------------------------------------------------------
// Kernel 1: U[k][b][:] = M1 @ interp(past,t_k)[b] + c1 for the 65 sample times,
// and z0[b] = M0 @ past[b,0] + c0 (k==0 has w==0, so P == past[:,0,:]).
// ---------------------------------------------------------------------------
__global__ void u_kernel(const float* __restrict__ past,
                         const float* __restrict__ M1, const float* __restrict__ c1,
                         const float* __restrict__ M0, const float* __restrict__ c0,
                         float* __restrict__ U, float* __restrict__ z0g) {
  extern __shared__ char smem[];
  float* M1s = (float*)smem;      // 4096
  float* c1s = M1s + 4096;        // 128
  float* M0s = c1s + 128;         // 4096
  float* c0s = M0s + 4096;        // 128
  float* P   = c0s + 128;         // 64 rows x 32

  const int k    = blockIdx.x;        // 0..64
  const int brow = blockIdx.y * 64;
  const int tid  = threadIdx.x;       // 256

  float pos = (float)k * (511.f / 64.f);
  int i0 = (int)floorf(pos);
  if (i0 < 0) i0 = 0; if (i0 > 511) i0 = 511;
  int i1 = i0 + 1; if (i1 > 511) i1 = 511;
  float w = pos - (float)i0;

  for (int i = tid; i < 4096; i += 256) { M1s[i] = M1[i]; M0s[i] = M0[i]; }
  if (tid < 128) { c1s[tid] = c1[tid]; c0s[tid] = c0[tid]; }
  for (int i = tid; i < 2048; i += 256) {
    int r = i >> 5, d = i & 31;
    size_t base = (size_t)(brow + r) * 512 * 32;
    P[i] = (1.f - w) * past[base + (size_t)i0 * 32 + d]
               + w  * past[base + (size_t)i1 * 32 + d];
  }
  __syncthreads();

  for (int e = 0; e < 32; ++e) {
    int idx = tid + e * 256;
    int r = idx >> 7, n = idx & 127;
    const float* mrow = M1s + n * 32;
    const float* prow = P + r * 32;
    float s = c1s[n];
    #pragma unroll
    for (int d = 0; d < 32; ++d) s += mrow[d] * prow[d];
    U[((size_t)k * BTOT + brow + r) * 128 + n] = s;
    if (k == 0) {
      const float* m0r = M0s + n * 32;
      float s0 = c0s[n];
      #pragma unroll
      for (int d = 0; d < 32; ++d) s0 += m0r[d] * prow[d];
      z0g[(size_t)(brow + r) * 128 + n] = s0;
    }
  }
}

// ---------------------------------------------------------------------------
// Persistent RK4 ODE kernel. Block = 128 threads (4 waves); each wave owns 16
// rows of the batch through all 32 RK4 steps.  Weights (W1z,w2,w3) live in LDS
// in bf16 WMMA B-operand tile layout; one odef eval = 3 x (16x128x128) GEMMs.
// ---------------------------------------------------------------------------
struct OdeCtx {
  const unsigned int* wt;   // LDS: 3 x 8192 words of packed B tiles
  const float* consts;      // LDS: ln_g | ln_b | b2 | b3 (4x128)
  float* H;                 // wave LDS 16x128 f32
  float* ST;                // wave LDS 32 f32 (mu, rsig)
  const float* U;
  int grow, m_l, half, ln;
};

__device__ __forceinline__ void odef_eval(const OdeCtx& cx, const float* I, int tk) {
  const int m_l = cx.m_l, half = cx.half, ln = cx.ln;
  // ---- layernorm stats (lanes 0..15 reduce one row each) ----
  if (ln < 16) {
    const float* row = I + ln * 128;
    float s = 0.f, ss = 0.f;
    for (int j = 0; j < 128; ++j) { float v = row[j]; s += v; ss += v * v; }
    float mu = s * (1.f / 128.f);
    float var = ss * (1.f / 128.f) - mu * mu;
    cx.ST[ln] = mu;
    cx.ST[16 + ln] = rsqrtf(var + 1e-5f);
  }
  wave_lds_sync();
  float mu = cx.ST[m_l], rs = cx.ST[16 + m_l];

  v8f c[8];
  // ---- GEMM1: c = U[tk] (bias W1x@xt + b1 prefolded), A = layernorm(z) ----
  const float* Ub = cx.U + ((size_t)tk * BTOT + cx.grow) * 128;
  #pragma unroll
  for (int nt = 0; nt < 8; ++nt) {
    int n = m_l + 16 * nt;
    #pragma unroll
    for (int r = 0; r < 8; ++r) c[nt][r] = Ub[(size_t)(r + 8 * half) * 128 + n];
  }
  // prefetch next time-slice of U while the dependent GEMM chain runs
  if (tk < 64)
    __builtin_prefetch(cx.U + ((size_t)(tk + 1) * BTOT + cx.grow) * 128 + ln * 64, 0, 1);

  for (int kt = 0; kt < 4; ++kt) {
    u32x8 A;
    const float* row = I + m_l * 128;
    int kb = kt * 32 + half * 8;
    #pragma unroll
    for (int j = 0; j < 4; ++j) {
      int k0 = kb + 2 * j;
      float a0 = (row[k0] - mu) * rs * cx.consts[k0] + cx.consts[128 + k0];
      float a1 = (row[k0 + 1] - mu) * rs * cx.consts[k0 + 1] + cx.consts[128 + k0 + 1];
      A[j] = pk2(a0, a1);
      int k1 = kb + 16 + 2 * j;
      float b0 = (row[k1] - mu) * rs * cx.consts[k1] + cx.consts[128 + k1];
      float b1v = (row[k1 + 1] - mu) * rs * cx.consts[k1 + 1] + cx.consts[128 + k1 + 1];
      A[4 + j] = pk2(b0, b1v);
    }
    #pragma unroll
    for (int nt = 0; nt < 8; ++nt) {
      u32x8 Bv = load_btile(cx.wt + (nt * 4 + kt) * 256 + ln * 8);
      c[nt] = wmma_bf16(A, Bv, c[nt]);
    }
  }
  #pragma unroll
  for (int nt = 0; nt < 8; ++nt) {
    int n = m_l + 16 * nt;
    #pragma unroll
    for (int r = 0; r < 8; ++r) cx.H[(r + 8 * half) * 128 + n] = silu(c[nt][r]);
  }
  wave_lds_sync();

  // ---- GEMM2: h2 = silu(h1 @ w2^T + b2) ----
  #pragma unroll
  for (int nt = 0; nt < 8; ++nt) {
    float bb = cx.consts[256 + m_l + 16 * nt];
    #pragma unroll
    for (int r = 0; r < 8; ++r) c[nt][r] = bb;
  }
  for (int kt = 0; kt < 4; ++kt) {
    u32x8 A;
    const float* row = cx.H + m_l * 128;
    int kb = kt * 32 + half * 8;
    #pragma unroll
    for (int j = 0; j < 4; ++j) {
      A[j]     = pk2(row[kb + 2 * j], row[kb + 2 * j + 1]);
      A[4 + j] = pk2(row[kb + 16 + 2 * j], row[kb + 16 + 2 * j + 1]);
    }
    #pragma unroll
    for (int nt = 0; nt < 8; ++nt) {
      u32x8 Bv = load_btile(cx.wt + 8192 + (nt * 4 + kt) * 256 + ln * 8);
      c[nt] = wmma_bf16(A, Bv, c[nt]);
    }
  }
  #pragma unroll
  for (int nt = 0; nt < 8; ++nt) {
    int n = m_l + 16 * nt;
    #pragma unroll
    for (int r = 0; r < 8; ++r) cx.H[(r + 8 * half) * 128 + n] = silu(c[nt][r]);
  }
  wave_lds_sync();

  // ---- GEMM3: k = h2 @ w3^T + b3  (no activation) ----
  #pragma unroll
  for (int nt = 0; nt < 8; ++nt) {
    float bb = cx.consts[384 + m_l + 16 * nt];
    #pragma unroll
    for (int r = 0; r < 8; ++r) c[nt][r] = bb;
  }
  for (int kt = 0; kt < 4; ++kt) {
    u32x8 A;
    const float* row = cx.H + m_l * 128;
    int kb = kt * 32 + half * 8;
    #pragma unroll
    for (int j = 0; j < 4; ++j) {
      A[j]     = pk2(row[kb + 2 * j], row[kb + 2 * j + 1]);
      A[4 + j] = pk2(row[kb + 16 + 2 * j], row[kb + 16 + 2 * j + 1]);
    }
    #pragma unroll
    for (int nt = 0; nt < 8; ++nt) {
      u32x8 Bv = load_btile(cx.wt + 16384 + (nt * 4 + kt) * 256 + ln * 8);
      c[nt] = wmma_bf16(A, Bv, c[nt]);
    }
  }
  #pragma unroll
  for (int nt = 0; nt < 8; ++nt) {
    int n = m_l + 16 * nt;
    #pragma unroll
    for (int r = 0; r < 8; ++r) cx.H[(r + 8 * half) * 128 + n] = c[nt][r];
  }
  wave_lds_sync();
}

__global__ __launch_bounds__(128, 1)
void ode_kernel(const float* __restrict__ w1, const float* __restrict__ w2g,
                const float* __restrict__ w3g, const float* __restrict__ b2g,
                const float* __restrict__ b3g, const float* __restrict__ lng,
                const float* __restrict__ lnb, const float* __restrict__ U,
                const float* __restrict__ z0g, float* __restrict__ condg) {
  extern __shared__ char smem[];
  unsigned int* wt = (unsigned int*)smem;            // 3 * 8192 words (96 KB)
  float* consts    = (float*)(smem + 98304);         // 4 * 128 f32
  const int tid = threadIdx.x;
  const int wv = tid >> 5, ln = tid & 31;
  const int m_l = ln & 15, half = ln >> 4;

  // Pack W1z / w2 / w3 into bf16 WMMA B-operand tiles:
  // B tile (nt,kt): lane la word j holds K = kt*32 + (la<16?0:8) + 2j (+8 for j>=4).
  for (int widx = tid; widx < 8192; widx += 128) {
    int tile = widx >> 8;                // nt*4 + kt
    int rem = widx & 255;
    int la = rem >> 3, j = rem & 7;
    int nt = tile >> 2, kt = tile & 3;
    int n = nt * 16 + (la & 15);
    int kb = kt * 32 + ((la >> 4) ? 8 : 0) + 2 * j + ((j >> 2) << 3);
    wt[widx]         = pk2(w1[n * 256 + kb],  w1[n * 256 + kb + 1]);   // W1z = w1[:,0:128]
    wt[8192 + widx]  = pk2(w2g[n * 128 + kb], w2g[n * 128 + kb + 1]);
    wt[16384 + widx] = pk2(w3g[n * 128 + kb], w3g[n * 128 + kb + 1]);
  }
  if (tid < 128) {
    consts[tid]       = lng[tid];
    consts[128 + tid] = lnb[tid];
    consts[256 + tid] = b2g[tid];
    consts[384 + tid] = b3g[tid];
  }
  __syncthreads();

  float* wbase = (float*)(smem + 100352) + wv * 8224;
  float* Z   = wbase;             // 16x128 state
  float* ZT  = wbase + 2048;      // RK4 trial state
  float* ACC = wbase + 4096;      // k accumulator
  float* H   = wbase + 6144;      // GEMM staging / odef output
  float* ST  = wbase + 8192;      // layernorm stats

  const int grow = blockIdx.x * 64 + wv * 16;
  for (int i = 0; i < 64; ++i)
    Z[i * 32 + ln] = z0g[(size_t)grow * 128 + i * 32 + ln];
  wave_lds_sync();

  OdeCtx cx{wt, consts, H, ST, U, grow, m_l, half, ln};
  const float dt = 1.f / 32.f;

  for (int stp = 0; stp < 32; ++stp) {
    odef_eval(cx, Z, 2 * stp);                        // k1
    for (int i = 0; i < 64; ++i) {
      int e = i * 32 + ln; float k = H[e];
      ACC[e] = k; ZT[e] = Z[e] + 0.5f * dt * k;
    }
    wave_lds_sync();
    odef_eval(cx, ZT, 2 * stp + 1);                   // k2
    for (int i = 0; i < 64; ++i) {
      int e = i * 32 + ln; float k = H[e];
      ACC[e] += 2.f * k; ZT[e] = Z[e] + 0.5f * dt * k;
    }
    wave_lds_sync();
    odef_eval(cx, ZT, 2 * stp + 1);                   // k3
    for (int i = 0; i < 64; ++i) {
      int e = i * 32 + ln; float k = H[e];
      ACC[e] += 2.f * k; ZT[e] = Z[e] + dt * k;
    }
    wave_lds_sync();
    odef_eval(cx, ZT, 2 * stp + 2);                   // k4
    for (int i = 0; i < 64; ++i) {
      int e = i * 32 + ln; float k = H[e];
      Z[e] = Z[e] + (dt / 6.f) * (ACC[e] + k);
    }
    wave_lds_sync();
  }
  for (int i = 0; i < 64; ++i)
    condg[(size_t)grow * 128 + i * 32 + ln] = Z[i * 32 + ln];
}

// ---------------------------------------------------------------------------
// Sinusoidal time embedding (B x 128).
// ---------------------------------------------------------------------------
__global__ void temb_kernel(const int* __restrict__ t, float* __restrict__ temb) {
  int idx = blockIdx.x * 256 + threadIdx.x;
  if (idx >= BTOT * 128) return;
  int b = idx >> 7, j = idx & 127;
  int jj = j & 63;
  float freq = __expf(-logf(10000.f) * (float)jj / 63.f);
  float arg = (float)t[b] * freq;
  temb[idx] = (j < 64) ? sinf(arg) : cosf(arg);
}

// ---------------------------------------------------------------------------
// Final MLP: h=concat(x_flat,cond,temb)(K=3328) -> 256 -> 256 -> 3072.
// Block = 128 threads (4 waves, 64 rows).  Cooperative bf16 B-slice staging.
// ---------------------------------------------------------------------------
__global__ __launch_bounds__(128, 1)
void mlp_kernel(const float* __restrict__ x_t, const float* __restrict__ condg,
                const float* __restrict__ tembg,
                const float* __restrict__ fw1, const float* __restrict__ fb1,
                const float* __restrict__ fw2, const float* __restrict__ fb2,
                const float* __restrict__ fw3, const float* __restrict__ fb3,
                float* __restrict__ out) {
  extern __shared__ char smem[];
  unsigned int* bs = (unsigned int*)smem;            // 4096-word B slice (N=256, K=32)
  float* Hrow = (float*)(smem + 16384);              // 4 waves * 16x256 f32
  const int tid = threadIdx.x;
  const int wv = tid >> 5, ln = tid & 31;
  const int m_l = ln & 15, half = ln >> 4;
  const int grow = blockIdx.x * 64 + wv * 16;
  float* Hw = Hrow + wv * 4096;

  v8f c[16];

  // ================= G1: (64 x 3328) @ fw1^T -> 256 =================
  #pragma unroll
  for (int nt = 0; nt < 16; ++nt) {
    float bb = fb1[nt * 16 + m_l];
    #pragma unroll
    for (int r = 0; r < 8; ++r) c[nt][r] = bb;
  }
  for (int kt = 0; kt < 104; ++kt) {
    __syncthreads();
    for (int widx = tid; widx < 4096; widx += 128) {
      int nt = widx >> 8, rem = widx & 255;
      int la = rem >> 3, j = rem & 7;
      int n = nt * 16 + (la & 15);
      int kb = kt * 32 + ((la >> 4) ? 8 : 0) + 2 * j + ((j >> 2) << 3);
      bs[widx] = pk2(fw1[(size_t)n * 3328 + kb], fw1[(size_t)n * 3328 + kb + 1]);
    }
    __syncthreads();
    const float* src;
    int koff;
    int kbase = kt * 32;
    if (kbase < 3072)      { src = x_t   + (size_t)(grow + m_l) * 3072; koff = kbase; }
    else if (kbase < 3200) { src = condg + (size_t)(grow + m_l) * 128;  koff = kbase - 3072; }
    else                   { src = tembg + (size_t)(grow + m_l) * 128;  koff = kbase - 3200; }
    u32x8 A;
    int kb = koff + half * 8;
    #pragma unroll
    for (int j = 0; j < 4; ++j) {
      A[j]     = pk2(src[kb + 2 * j], src[kb + 2 * j + 1]);
      A[4 + j] = pk2(src[kb + 16 + 2 * j], src[kb + 16 + 2 * j + 1]);
    }
    #pragma unroll
    for (int nt = 0; nt < 16; ++nt) {
      u32x8 Bv = load_btile(bs + nt * 256 + ln * 8);
      c[nt] = wmma_bf16(A, Bv, c[nt]);
    }
  }
  __syncthreads();
  #pragma unroll
  for (int nt = 0; nt < 16; ++nt) {
    int n = m_l + 16 * nt;
    #pragma unroll
    for (int r = 0; r < 8; ++r) Hw[(r + 8 * half) * 256 + n] = silu(c[nt][r]);
  }
  __syncthreads();

  // ================= G2: (64 x 256) @ fw2^T -> 256 =================
  #pragma unroll
  for (int nt = 0; nt < 16; ++nt) {
    float bb = fb2[nt * 16 + m_l];
    #pragma unroll
    for (int r = 0; r < 8; ++r) c[nt][r] = bb;
  }
  for (int kt = 0; kt < 8; ++kt) {
    __syncthreads();
    for (int widx = tid; widx < 4096; widx += 128) {
      int nt = widx >> 8, rem = widx & 255;
      int la = rem >> 3, j = rem & 7;
      int n = nt * 16 + (la & 15);
      int kb = kt * 32 + ((la >> 4) ? 8 : 0) + 2 * j + ((j >> 2) << 3);
      bs[widx] = pk2(fw2[(size_t)n * 256 + kb], fw2[(size_t)n * 256 + kb + 1]);
    }
    __syncthreads();
    const float* row = Hw + m_l * 256;
    int kb = kt * 32 + half * 8;
    u32x8 A;
    #pragma unroll
    for (int j = 0; j < 4; ++j) {
      A[j]     = pk2(row[kb + 2 * j], row[kb + 2 * j + 1]);
      A[4 + j] = pk2(row[kb + 16 + 2 * j], row[kb + 16 + 2 * j + 1]);
    }
    #pragma unroll
    for (int nt = 0; nt < 16; ++nt) {
      u32x8 Bv = load_btile(bs + nt * 256 + ln * 8);
      c[nt] = wmma_bf16(A, Bv, c[nt]);
    }
  }
  __syncthreads();
  #pragma unroll
  for (int nt = 0; nt < 16; ++nt) {
    int n = m_l + 16 * nt;
    #pragma unroll
    for (int r = 0; r < 8; ++r) Hw[(r + 8 * half) * 256 + n] = silu(c[nt][r]);
  }
  __syncthreads();

  // ================= G3: (64 x 256) @ fw3^T -> 3072, N tiled by 256 =========
  for (int nc = 0; nc < 12; ++nc) {
    #pragma unroll
    for (int nt = 0; nt < 16; ++nt) {
      float bb = fb3[nc * 256 + nt * 16 + m_l];
      #pragma unroll
      for (int r = 0; r < 8; ++r) c[nt][r] = bb;
    }
    for (int kt = 0; kt < 8; ++kt) {
      __syncthreads();
      for (int widx = tid; widx < 4096; widx += 128) {
        int nt = widx >> 8, rem = widx & 255;
        int la = rem >> 3, j = rem & 7;
        int n = nc * 256 + nt * 16 + (la & 15);
        int kb = kt * 32 + ((la >> 4) ? 8 : 0) + 2 * j + ((j >> 2) << 3);
        bs[widx] = pk2(fw3[(size_t)n * 256 + kb], fw3[(size_t)n * 256 + kb + 1]);
      }
      __syncthreads();
      const float* row = Hw + m_l * 256;
      int kb = kt * 32 + half * 8;
      u32x8 A;
      #pragma unroll
      for (int j = 0; j < 4; ++j) {
        A[j]     = pk2(row[kb + 2 * j], row[kb + 2 * j + 1]);
        A[4 + j] = pk2(row[kb + 16 + 2 * j], row[kb + 16 + 2 * j + 1]);
      }
      #pragma unroll
      for (int nt = 0; nt < 16; ++nt) {
        u32x8 Bv = load_btile(bs + nt * 256 + ln * 8);
        c[nt] = wmma_bf16(A, Bv, c[nt]);
      }
    }
    #pragma unroll
    for (int nt = 0; nt < 16; ++nt) {
      int n = nc * 256 + nt * 16 + m_l;
      #pragma unroll
      for (int r = 0; r < 8; ++r)
        out[(size_t)(grow + r + 8 * half) * 3072 + n] = c[nt][r];
    }
  }
}

// ---------------------------------------------------------------------------
extern "C" void kernel_launch(void* const* d_in, const int* in_sizes, int n_in,
                              void* d_out, int out_size, void* d_ws, size_t ws_size,
                              hipStream_t stream) {
  const float* x_t  = (const float*)d_in[0];
  const float* past = (const float*)d_in[1];
  const int*   tt   = (const int*)d_in[2];
  const float* xw   = (const float*)d_in[3];
  const float* xb   = (const float*)d_in[4];
  const float* z0w  = (const float*)d_in[5];
  const float* z0b  = (const float*)d_in[6];
  const float* lng  = (const float*)d_in[7];
  const float* lnb  = (const float*)d_in[8];
  const float* w1   = (const float*)d_in[9];
  const float* b1   = (const float*)d_in[10];
  const float* w2   = (const float*)d_in[11];
  const float* b2   = (const float*)d_in[12];
  const float* w3   = (const float*)d_in[13];
  const float* b3   = (const float*)d_in[14];
  const float* fw1  = (const float*)d_in[15];
  const float* fb1  = (const float*)d_in[16];
  const float* fw2  = (const float*)d_in[17];
  const float* fb2  = (const float*)d_in[18];
  const float* fw3  = (const float*)d_in[19];
  const float* fb3  = (const float*)d_in[20];
  float* out = (float*)d_out;

  float* ws = (float*)d_ws;
  float* M1    = ws;                       // 4096
  float* c1    = ws + 4096;                // 128
  float* M0    = ws + 4224;                // 4096
  float* c0    = ws + 8320;                // 128
  float* U     = ws + 8448;                // 65*2048*128
  float* z0g   = U + (size_t)65 * BTOT * 128;
  float* condg = z0g + (size_t)BTOT * 128;
  float* tembg = condg + (size_t)BTOT * 128;

  prep_kernel<<<1, 128, 0, stream>>>(w1, b1, xw, xb, z0w, z0b, M1, c1, M0, c0);

  dim3 ug(65, BTOT / 64);
  u_kernel<<<ug, 256, 41984, stream>>>(past, M1, c1, M0, c0, U, z0g);

  ode_kernel<<<BTOT / 64, 128, 231936, stream>>>(w1, w2, w3, b2, b3, lng, lnb,
                                                 U, z0g, condg);

  temb_kernel<<<(BTOT * 128 + 255) / 256, 256, 0, stream>>>(tt, tembg);

  mlp_kernel<<<BTOT / 64, 128, 81920, stream>>>(x_t, condg, tembg,
                                                fw1, fb1, fw2, fb2, fw3, fb3, out);

  (void)in_sizes; (void)n_in; (void)out_size; (void)ws_size;
}
